// MsgGraphNetHard_63917703299078
// MI455X (gfx1250) — matrix operand order repaired and checked by reading, not compiled, not run on hardware
//
#include <hip/hip_runtime.h>
#include <hip/hip_bf16.h>

typedef __attribute__((ext_vector_type(16))) _Float16 v16h;
typedef __attribute__((ext_vector_type(8)))  _Float16 v8h;
typedef __attribute__((ext_vector_type(4)))  _Float16 v4h;
typedef __attribute__((ext_vector_type(8)))  float    v8f;

#define XSTR 136   // xs row stride (halves); rows indexed [b*3 + agent], 48 rows of 128
#define HSTR 264   // he/agg/h2 row stride (halves); 256 data + 8 pad

// f16 weight workspace offsets (halves)
#define OFF_FE1 0
#define OFF_FV1 (OFF_FE1 + 256 * 256)
#define OFF_FE2 (OFF_FV1 + 256 * 384)
#define OFF_FV2 (OFF_FE2 + 256 * 512)
#define WS_HALVES (OFF_FV2 + 256 * 384)

// SRC = [0,0,1,1,2,2], DST = [1,2,0,2,0,1]
__device__ __forceinline__ int srcOf(int e) { return e >> 1; }
__device__ __forceinline__ int dstOf(int e) {
    return (e == 0) ? 1 : (e == 1) ? 2 : (e == 2) ? 0 : (e == 3) ? 2 : (e == 4) ? 0 : 1;
}

// ---- WMMA fragment helpers -------------------------------------------------

// 16x32 f16 A-fragment from LDS (row-major activations).
// Lanes 0-15 hold K {0..7, 16..23} of row M=lane; lanes 16-31 hold K {8..15, 24..31}.
__device__ __forceinline__ v16h a_frag(const _Float16* base, int rowStride, int col, int lane) {
    const _Float16* p = base + (lane & 15) * rowStride + col + ((lane & 16) ? 8 : 0);
    v8h lo = *(const v8h*)p;
    v8h hi = *(const v8h*)(p + 16);
    v16h r;
#pragma unroll
    for (int i = 0; i < 8; ++i) { r[i] = lo[i]; r[8 + i] = hi[i]; }
    return r;
}

// B-fragment straight from global (weights L2-resident):
// column n = weight row (output feature); lanes 0-15 hold K {0..7, 16..23}, lanes 16-31 hold K {8..15, 24..31}.
__device__ __forceinline__ v16h b_frag(const _Float16* __restrict__ W, int Krow,
                                       int nbase, int kf, int lane) {
    const _Float16* p =
        W + (size_t)(nbase + (lane & 15)) * Krow + kf * 32 + ((lane & 16) ? 8 : 0);
    v8h lo = *(const v8h*)p;
    v8h hi = *(const v8h*)(p + 16);
    v16h r;
#pragma unroll
    for (int i = 0; i < 8; ++i) { r[i] = lo[i]; r[8 + i] = hi[i]; }
    return r;
}

// ReLU + f32->f16 store of one 16x16 C tile into LDS.
// C layout: lane -> N = lane&15, VGPR v -> M = v + (lane>=16 ? 8 : 0).
__device__ __forceinline__ void store_tile(_Float16* __restrict__ outLds, int rowBase,
                                           int nbase, int lane, const v8f& acc) {
    const int col = nbase + (lane & 15);
    const int r0 = rowBase + ((lane & 16) ? 8 : 0);
#pragma unroll
    for (int v = 0; v < 8; ++v) {
        const float xv = acc[v];
        outLds[(r0 + v) * HSTR + col] = (_Float16)(xv > 0.f ? xv : 0.f);
    }
}

// Weight-stationary GEMM stage with dual-n-tile A-reuse: each wave owns n-tiles
// {2w, 2w+1}; per k-step it loads 2 B-fragments (L2) + 1 A-fragment (LDS) per m
// and issues TWO wmma per A read (1 ds_b128 : 1 wmma).
template <int MT, int CH, int KF, typename AF>
__device__ __forceinline__ void gemm_stage(const _Float16* __restrict__ W, int Krow,
                                           const float* __restrict__ bias,
                                           _Float16* __restrict__ outLds,
                                           int wave, int lane, AF&& aload) {
    const int nb0 = (wave * 2) * 16;
    const int nb1 = nb0 + 16;
    const float bv0 = bias[nb0 + (lane & 15)];
    const float bv1 = bias[nb1 + (lane & 15)];
#pragma unroll
    for (int mc = 0; mc < MT / CH; ++mc) {
        v8f acc0[CH], acc1[CH];
#pragma unroll
        for (int m = 0; m < CH; ++m) {
#pragma unroll
            for (int i = 0; i < 8; ++i) { acc0[m][i] = bv0; acc1[m][i] = bv1; }
        }
#pragma unroll
        for (int kf = 0; kf < KF; ++kf) {
            const v16h bf0 = b_frag(W, Krow, nb0, kf, lane);
            const v16h bf1 = b_frag(W, Krow, nb1, kf, lane);
#pragma unroll
            for (int m = 0; m < CH; ++m) {
                const v16h af = aload(mc * CH + m, kf);
                acc0[m] = __builtin_amdgcn_wmma_f32_16x16x32_f16(
                    false, af, false, bf0, (short)0, acc0[m], false, false);
                acc1[m] = __builtin_amdgcn_wmma_f32_16x16x32_f16(
                    false, af, false, bf1, (short)0, acc1[m], false, false);
            }
        }
#pragma unroll
        for (int m = 0; m < CH; ++m) {
            store_tile(outLds, (mc * CH + m) * 16, nb0, lane, acc0[m]);
            store_tile(outLds, (mc * CH + m) * 16, nb1, lane, acc1[m]);
        }
    }
}

// mean of the 2 in-edges per node: agg[n*16+b] = 0.5*(he[e1*16+b] + he[e2*16+b])
__device__ __forceinline__ void aggregate(const _Float16* __restrict__ he,
                                          _Float16* __restrict__ agg, int tid) {
    for (int i = tid; i < 48 * 32; i += 256) {
        const int row = i >> 5;            // n*16 + b
        const int n = row >> 4, b = row & 15;
        const int col = (i & 31) * 8;
        const int e1 = (n == 0) ? 2 : (n == 1) ? 0 : 1;
        const int e2 = (n == 0) ? 4 : (n == 1) ? 5 : 3;
        v8h a = *(const v8h*)(he + (e1 * 16 + b) * HSTR + col);
        v8h c = *(const v8h*)(he + (e2 * 16 + b) * HSTR + col);
        v8h s;
#pragma unroll
        for (int j = 0; j < 8; ++j) s[j] = (_Float16)((a[j] + c[j]) * (_Float16)0.5f);
        *(v8h*)(agg + row * HSTR + col) = s;
    }
}

// ---- weight f32 -> f16 pre-pass --------------------------------------------
__global__ void cvt_w_kernel(const float* __restrict__ s, _Float16* __restrict__ d, int n) {
    int i = (blockIdx.x * 256 + threadIdx.x) * 4;
    if (i + 3 < n) {
        const float4 v = *(const float4*)(s + i);
        v4h h;
        h[0] = (_Float16)v.x; h[1] = (_Float16)v.y;
        h[2] = (_Float16)v.z; h[3] = (_Float16)v.w;
        *(v4h*)(d + i) = h;
    }
}

// ---- fused graph-net kernel: one workgroup = 16 batch elements --------------
__global__ __launch_bounds__(256) void mgn_kernel(
    const float* __restrict__ xg,
    const float* __restrict__ fe1_b, const float* __restrict__ fv1_b,
    const float* __restrict__ fe2_b, const float* __restrict__ fv2_b,
    const float* __restrict__ Vw, const float* __restrict__ Vb,
    const _Float16* __restrict__ ws, float* __restrict__ out) {
    extern __shared__ _Float16 smem[];
    _Float16* xs  = smem;               // 48 x XSTR  (x tile, f16)
    _Float16* he  = xs + 48 * XSTR;     // 96 x HSTR  (edge features, rows e*16+b)
    _Float16* agg = he + 96 * HSTR;     // 48 x HSTR  (aggregated, rows n*16+b)
    _Float16* h2  = agg + 48 * HSTR;    // 48 x HSTR  (node features / h3)
    float* red    = (float*)(h2 + 48 * HSTR);  // 16 x 16 f32 partials

    const int tid  = threadIdx.x;
    const int lane = tid & 31;
    const int wave = tid >> 5;
    const int b0   = blockIdx.x * 16;

    // ---- load x tile, convert to f16 into LDS (rows = b*3 + agent) ----
    {
        const float* xt = xg + (size_t)b0 * 384;
        for (int c = tid; c < 1536; c += 256) {
            const int row = c >> 5, col = (c & 31) << 2;
            const float4 v = *(const float4*)(xt + row * 128 + col);
            v4h h;
            h[0] = (_Float16)v.x; h[1] = (_Float16)v.y;
            h[2] = (_Float16)v.z; h[3] = (_Float16)v.w;
            *(v4h*)(xs + row * XSTR + col) = h;
        }
    }
    // warm L2 with the next block's x tile (blocks launch roughly in order);
    // no live state carried -> zero register-pressure cost.
    if (blockIdx.x + 1 < gridDim.x && tid < 192) {
        const char* nx = (const char*)(xg + ((size_t)b0 + 16) * 384);
        __builtin_prefetch(nx + tid * 128, 0, 1);
    }
    __syncthreads();

    // ---- edge layer 1: in = cat(x[src], x[dst]) (K=256) ----
    {
        auto af = [&](int e, int kf) {
            const int a = (kf < 4) ? srcOf(e) : dstOf(e);
            return a_frag(xs + a * XSTR, 3 * XSTR, (kf & 3) * 32, lane);
        };
        gemm_stage<6, 3, 8>(ws + OFF_FE1, 256, fe1_b, he, wave, lane, af);
    }
    __syncthreads();
    aggregate(he, agg, tid);
    __syncthreads();

    // ---- node layer 1: in = cat(agg, x) (K=384) ----
    {
        auto af = [&](int n, int kf) {
            return (kf < 8) ? a_frag(agg + n * 16 * HSTR, HSTR, kf * 32, lane)
                            : a_frag(xs + n * XSTR, 3 * XSTR, (kf - 8) * 32, lane);
        };
        gemm_stage<3, 3, 12>(ws + OFF_FV1, 384, fv1_b, h2, wave, lane, af);
    }
    __syncthreads();

    // ---- edge layer 2: in = cat(h2[src], h2[dst]) (K=512) ----
    {
        auto af = [&](int e, int kf) {
            const int a = (kf < 8) ? srcOf(e) : dstOf(e);
            return a_frag(h2 + a * 16 * HSTR, HSTR, (kf & 7) * 32, lane);
        };
        gemm_stage<6, 3, 16>(ws + OFF_FE2, 512, fe2_b, he, wave, lane, af);
    }
    __syncthreads();
    aggregate(he, agg, tid);
    __syncthreads();

    // ---- node layer 2: in = cat(agg, x) (K=384), result h3 -> h2 buffer ----
    {
        auto af = [&](int n, int kf) {
            return (kf < 8) ? a_frag(agg + n * 16 * HSTR, HSTR, kf * 32, lane)
                            : a_frag(xs + n * XSTR, 3 * XSTR, (kf - 8) * 32, lane);
        };
        gemm_stage<3, 3, 12>(ws + OFF_FV2, 384, fv2_b, h2, wave, lane, af);
    }
    __syncthreads();

    // ---- max over agents + value head ----
    {
        const int b = tid >> 4, seg = tid & 15;
        float p = 0.f;
#pragma unroll
        for (int j = 0; j < 16; ++j) {
            const int col = seg * 16 + j;
            const float m0 = (float)h2[b * HSTR + col];
            const float m1 = (float)h2[(16 + b) * HSTR + col];
            const float m2 = (float)h2[(32 + b) * HSTR + col];
            p += fmaxf(m0, fmaxf(m1, m2)) * Vw[col];
        }
        red[b * 16 + seg] = p;
    }
    __syncthreads();
    if (tid < 16) {
        float s = Vb[0];
#pragma unroll
        for (int j = 0; j < 16; ++j) s += red[tid * 16 + j];
        out[b0 + tid] = s;
    }
}

extern "C" void kernel_launch(void* const* d_in, const int* in_sizes, int n_in,
                              void* d_out, int out_size, void* d_ws, size_t ws_size,
                              hipStream_t stream) {
    (void)n_in; (void)out_size; (void)ws_size;
    const float* x     = (const float*)d_in[0];
    const float* fe1_w = (const float*)d_in[1];
    const float* fe1_b = (const float*)d_in[2];
    const float* fv1_w = (const float*)d_in[3];
    const float* fv1_b = (const float*)d_in[4];
    const float* fe2_w = (const float*)d_in[5];
    const float* fe2_b = (const float*)d_in[6];
    const float* fv2_w = (const float*)d_in[7];
    const float* fv2_b = (const float*)d_in[8];
    const float* Vw    = (const float*)d_in[9];
    const float* Vb    = (const float*)d_in[10];
    float* out   = (float*)d_out;
    _Float16* ws = (_Float16*)d_ws;

    // f32 -> f16 weight pre-pass (weights stay L2-resident, ~768 KB)
    cvt_w_kernel<<<64,  256, 0, stream>>>(fe1_w, ws + OFF_FE1, 256 * 256);
    cvt_w_kernel<<<96,  256, 0, stream>>>(fv1_w, ws + OFF_FV1, 256 * 384);
    cvt_w_kernel<<<128, 256, 0, stream>>>(fe2_w, ws + OFF_FE2, 256 * 512);
    cvt_w_kernel<<<96,  256, 0, stream>>>(fv2_w, ws + OFF_FV2, 256 * 384);

    const int B = in_sizes[0] / 384;   // 65536
    const int nblk = B / 16;
    const size_t smem =
        (size_t)(48 * XSTR + 96 * HSTR + 48 * HSTR + 48 * HSTR) * sizeof(_Float16) +
        16 * 16 * sizeof(float);
    hipFuncSetAttribute(reinterpret_cast<const void*>(mgn_kernel),
                        hipFuncAttributeMaxDynamicSharedMemorySize, (int)smem);
    mgn_kernel<<<nblk, 256, smem, stream>>>(x, fe1_b, fv1_b, fe2_b, fv2_b, Vw, Vb, ws, out);
}
